// MSRAttention_63239098466541
// MI455X (gfx1250) — matrix operand order repaired and checked by reading, not compiled, not run on hardware
//
#include <hip/hip_runtime.h>
#include <hip/hip_bf16.h>

typedef __attribute__((ext_vector_type(16))) _Float16 v16h;
typedef __attribute__((ext_vector_type(8)))  float    v8f;

#define WMMA_F16(a,b,c) __builtin_amdgcn_wmma_f32_16x16x32_f16(false,(a),false,(b),(short)0,(c),false,false)

static constexpr int BATCH = 2;
static constexpr int H_ = 112, W_ = 112;
static constexpr int N_ = H_ * W_;          // 12544
static constexpr int DIM = 512;
static constexpr int BN = BATCH * N_;       // 25088
static constexpr int D0 = 64;               // global branch dim
static constexpr int DL = 448;              // local dims total
static constexpr int NHD = 8;               // heads
static constexpr int HD = 28;               // head dim
static constexpr int HS = 56;               // H/SR
static constexpr int NR = HS * HS;          // 3136
static constexpr int NRT16 = NR / 16;       // 196
static constexpr int NRT32 = NR / 32;       // 98

union frag_u { v16h v; uint4 q[2]; };

// ---------------- CDNA5 async global->LDS DMA (no data VGPRs; ASYNCcnt) ----------------
__device__ __forceinline__ void async_copy_b128(unsigned lds_byte_addr, const void* gaddr) {
  asm volatile("global_load_async_to_lds_b128 %0, %1, off"
               :: "v"(lds_byte_addr), "v"(gaddr)
               : "memory");
}
__device__ __forceinline__ void wait_async() {
  asm volatile("s_wait_asynccnt 0x0" ::: "memory");
}

// ---------------- WMMA fragment loaders: all b128 LDS traffic ----------------
__device__ __forceinline__ v16h load_a_frag(const _Float16* t, int ld, int lane, int koff) {
  int m  = lane & 15;
  int kb = (lane & 16) ? 8 : 0;
  const uint4* p = (const uint4*)(t + m * ld + koff + kb);
  frag_u f;
  f.q[0] = p[0];   // halves 0..7   -> K = kb .. kb+7
  f.q[1] = p[2];   // halves 8..15  -> K = 16+kb .. 16+kb+7  (+16 halves)
  return f.v;
}
__device__ __forceinline__ v16h load_bswz_frag(const _Float16* t, int lane, int blk) {
  int n  = lane & 15;
  int kb = (lane & 16) ? 8 : 0;
  const uint4* p = (const uint4*)(t + blk * 512 + n * 32 + kb);
  frag_u f;
  f.q[0] = p[0];
  f.q[1] = p[2];
  return f.v;
}

// ---------------- converts ----------------
__global__ void k_cvt(const float* __restrict__ in, _Float16* __restrict__ outp, long n) {
  long i = (long)blockIdx.x * 256 + threadIdx.x;
  if (i < n) outp[i] = (_Float16)in[i];
}
// weight (logical KxN; trans!=0: input stored NxK) -> swizzled B tiles:
// tile (kt,nt) at ((kt*(N/16))+nt)*512, element [n*32+kk] = B[kt*32+kk][nt*16+n]
__global__ void k_cvt_swz(const float* __restrict__ in, _Float16* __restrict__ outp,
                          int K, int N, int trans) {
  long i = (long)blockIdx.x * 256 + threadIdx.x;
  if (i >= (long)K * N) return;
  int half = (int)(i & 511);
  long tile = i >> 9;
  int ntiles = N >> 4;
  int kt = (int)(tile / ntiles), nt = (int)(tile % ntiles);
  int n = half >> 5, kk = half & 31;
  int k = kt * 32 + kk, col = nt * 16 + n;
  float v = trans ? in[(size_t)col * K + k] : in[(size_t)k * N + col];
  outp[i] = (_Float16)v;
}

// --- WMMA GEMM: C(MxN)=A(MxK)@B(KxN)+bias; 32x64 C/wave, async-DMA LDS double-buffer ---
__global__ void k_gemm_wmma(const _Float16* __restrict__ A, const _Float16* __restrict__ Bsw,
                            const float* __restrict__ bias, float* __restrict__ C,
                            int M, int N, int K) {
  __shared__ __align__(16) _Float16 As[2][32 * 32];
  __shared__ __align__(16) _Float16 Bs[2][4 * 512];
  int lane = threadIdx.x;
  int row0 = blockIdx.x * 32, col0 = blockIdx.y * 64;
  int ntiles = N >> 4;
  const v8f vz = {0.f,0.f,0.f,0.f,0.f,0.f,0.f,0.f};
  v8f acc[2][4] = {{vz,vz,vz,vz},{vz,vz,vz,vz}};
  auto issue_stage = [&](int k0, int bufsel) {
    // A tile: 32x32 halves = 128 uint4 -> 4 per lane
#pragma unroll
    for (int it = 0; it < 4; ++it) {
      int i = lane + it * 32;
      int r = i >> 2, seg = i & 3;
      async_copy_b128((unsigned)(size_t)&As[bufsel][i * 8],
                      &A[(size_t)(row0 + r) * K + k0 + seg * 8]);
    }
    // B tiles: 4x512 halves = 256 uint4 -> 8 per lane (contiguous swizzled blocks)
    const _Float16* bsrc = &Bsw[((size_t)(k0 >> 5) * ntiles + (col0 >> 4)) * 512];
#pragma unroll
    for (int it = 0; it < 8; ++it) {
      int i = lane + it * 32;
      async_copy_b128((unsigned)(size_t)&Bs[bufsel][i * 8], bsrc + i * 8);
    }
  };
  issue_stage(0, 0);
  wait_async();
  __syncthreads();
  int buf = 0;
  for (int k0 = 0; k0 < K; k0 += 32) {
    bool more = (k0 + 32) < K;
    if (more) issue_stage(k0 + 32, buf ^ 1);   // DMA next tile while computing
    v16h a0 = load_a_frag(As[buf], 32, lane, 0);
    v16h a1 = load_a_frag(As[buf] + 16 * 32, 32, lane, 0);
#pragma unroll
    for (int j = 0; j < 4; ++j) {
      v16h b = load_bswz_frag(Bs[buf], lane, j);
      acc[0][j] = WMMA_F16(a0, b, acc[0][j]);
      acc[1][j] = WMMA_F16(a1, b, acc[1][j]);
    }
    if (more) {
      wait_async();
      __syncthreads();
      buf ^= 1;
    }
  }
  int nn = lane & 15, hi = lane >> 4;
#pragma unroll
  for (int rt = 0; rt < 2; ++rt)
#pragma unroll
    for (int j = 0; j < 4; ++j)
#pragma unroll
      for (int r = 0; r < 8; ++r) {
        int row = row0 + rt * 16 + r + 8 * hi, col = col0 + j * 16 + nn;
        float v = acc[rt][j][r] + (bias ? bias[col] : 0.f);
        C[(size_t)row * N + col] = v;
      }
}

// ---------------- global branch: stride-2 2x2 conv + LayerNorm -> f16 ----------------
__global__ void k_sr_ln(const float* __restrict__ x, const float* __restrict__ srw,
                        const float* __restrict__ srb, const float* __restrict__ g,
                        const float* __restrict__ bb, _Float16* __restrict__ outp) {
  __shared__ float sv[64];
  int c = threadIdx.x;
  int bid = blockIdx.x;
  int b = bid / NR, nr = bid % NR;
  int y = nr / HS, xx = nr % HS;
  float acc = srb[c];
  for (int ci = 0; ci < D0; ++ci) {
#pragma unroll
    for (int ky = 0; ky < 2; ++ky)
#pragma unroll
      for (int kx = 0; kx < 2; ++kx)
        acc += srw[((c * D0 + ci) * 2 + ky) * 2 + kx] *
               x[((size_t)b * N_ + (2 * y + ky) * W_ + 2 * xx + kx) * DIM + ci];
  }
  sv[c] = acc; __syncthreads();
  for (int s = 32; s > 0; s >>= 1) { if (c < s) sv[c] += sv[c + s]; __syncthreads(); }
  float m = sv[0] / 64.f; __syncthreads();
  float d = acc - m;
  sv[c] = d * d; __syncthreads();
  for (int s = 32; s > 0; s >>= 1) { if (c < s) sv[c] += sv[c + s]; __syncthreads(); }
  float var = sv[0] / 64.f;
  outp[(size_t)bid * 64 + c] = (_Float16)(d * rsqrtf(var + 1e-5f) * g[c] + bb[c]);
}

// ---------------- k_g/v_g finalize: split kvg, v += dw3x3+bias; emit swizzled ----------------
__global__ void k_finalize_kvg(const float* __restrict__ kvg, const float* __restrict__ lw,
                               const float* __restrict__ lb, _Float16* __restrict__ kf,
                               _Float16* __restrict__ vf) {
  int i = blockIdx.x * 256 + threadIdx.x;
  if (i >= BATCH * NR * D0) return;
  int c = i % D0; int t = i / D0;
  int nr = t % NR; int b = t / NR;
  int y = nr / HS, xx = nr % HS;
  float kval = kvg[((size_t)b * NR + nr) * 128 + c];
  float v = kvg[((size_t)b * NR + nr) * 128 + D0 + c];
  float acc = lb[c];
#pragma unroll
  for (int ky = 0; ky < 3; ++ky) {
    int yy = y + ky - 1; if (yy < 0 || yy >= HS) continue;
#pragma unroll
    for (int kx = 0; kx < 3; ++kx) {
      int xw = xx + kx - 1; if (xw < 0 || xw >= HS) continue;
      acc += kvg[((size_t)b * NR + yy * HS + xw) * 128 + D0 + c] * lw[c * 9 + ky * 3 + kx];
    }
  }
  int ktile = nr >> 4, n = nr & 15, dt = c >> 5, kk = c & 31;
  kf[(((size_t)b * NRT16 + ktile) * 2 + dt) * 512 + n * 32 + kk] = (_Float16)kval;
  int vc = nr >> 5, kk2 = nr & 31, j = c >> 4, n2 = c & 15;
  vf[(((size_t)b * NRT32 + vc) * 4 + j) * 512 + n2 * 32 + kk2] = (_Float16)(v + acc);
}

// ------- flash attention (global branch), 1 wave / 16 queries, async KV pipeline -------
__global__ void k_flash_global(const float* __restrict__ q, const _Float16* __restrict__ kf,
                               const _Float16* __restrict__ vf, _Float16* __restrict__ out16) {
  __shared__ __align__(16) _Float16 qs[16 * 64];
  __shared__ __align__(16) _Float16 ks[2][4 * 512];
  __shared__ __align__(16) _Float16 vs[2][4 * 512];
  __shared__ __align__(16) _Float16 ps[16 * 32];
  int lane = threadIdx.x;
  int tile = blockIdx.x;
  int ntile = N_ / 16;
  int b = tile / ntile, qt = tile % ntile;
  auto issue_kv = [&](int c0, int bufsel) {
    const _Float16* ksrc = &kf[(((size_t)b * NRT16 + (c0 >> 4)) * 2) * 512];
    const _Float16* vsrc = &vf[(((size_t)b * NRT32 + (c0 >> 5)) * 4) * 512];
#pragma unroll
    for (int it = 0; it < 8; ++it) {
      int i = lane + it * 32;
      async_copy_b128((unsigned)(size_t)&ks[bufsel][i * 8], ksrc + i * 8);
      async_copy_b128((unsigned)(size_t)&vs[bufsel][i * 8], vsrc + i * 8);
    }
  };
  issue_kv(0, 0);
#pragma unroll
  for (int it = 0; it < 32; ++it) {
    int i = lane + it * 32;
    int r = i >> 6, c = i & 63;
    qs[i] = (_Float16)(q[((size_t)b * N_ + qt * 16 + r) * DIM + c] * 0.125f); // 64^-0.5
  }
  wait_async();
  __syncthreads();
  v16h a0 = load_a_frag(qs, 64, lane, 0);
  v16h a1 = load_a_frag(qs, 64, lane, 32);
  const v8f vz = {0.f,0.f,0.f,0.f,0.f,0.f,0.f,0.f};
  v8f o[4] = {vz, vz, vz, vz};
  float rowm[8], rowl[8];
#pragma unroll
  for (int r = 0; r < 8; ++r) { rowm[r] = -1e30f; rowl[r] = 0.f; }
  int nn = lane & 15, hi = lane >> 4;
  int buf = 0;
  for (int c0 = 0; c0 < NR; c0 += 32) {
    bool more = (c0 + 32) < NR;
    if (more) issue_kv(c0 + 32, buf ^ 1);   // DMA next chunk during compute
    v8f s0 = vz, s1 = vz;
    s0 = WMMA_F16(a0, load_bswz_frag(ks[buf], lane, 0), s0);  // keys 0-15, dims 0-31
    s0 = WMMA_F16(a1, load_bswz_frag(ks[buf], lane, 1), s0);  // keys 0-15, dims 32-63
    s1 = WMMA_F16(a0, load_bswz_frag(ks[buf], lane, 2), s1);  // keys 16-31
    s1 = WMMA_F16(a1, load_bswz_frag(ks[buf], lane, 3), s1);
    float alpha[8];
#pragma unroll
    for (int r = 0; r < 8; ++r) {
      float mx = fmaxf(s0[r], s1[r]);
      for (int d = 1; d < 16; d <<= 1) mx = fmaxf(mx, __shfl_xor(mx, d, 16));
      float nm = fmaxf(rowm[r], mx);
      alpha[r] = __expf(rowm[r] - nm);
      rowm[r] = nm;
      float p0 = __expf(s0[r] - nm), p1 = __expf(s1[r] - nm);
      float rs = p0 + p1;
      for (int d = 1; d < 16; d <<= 1) rs += __shfl_xor(rs, d, 16);
      rowl[r] = rowl[r] * alpha[r] + rs;
      s0[r] = p0; s1[r] = p1;
    }
#pragma unroll
    for (int j = 0; j < 4; ++j)
#pragma unroll
      for (int r = 0; r < 8; ++r) o[j][r] *= alpha[r];
#pragma unroll
    for (int r = 0; r < 8; ++r) {
      ps[(r + 8 * hi) * 32 + nn]      = (_Float16)s0[r];
      ps[(r + 8 * hi) * 32 + 16 + nn] = (_Float16)s1[r];
    }
    __syncthreads();
    v16h pa = load_a_frag(ps, 32, lane, 0);
#pragma unroll
    for (int j = 0; j < 4; ++j)
      o[j] = WMMA_F16(pa, load_bswz_frag(vs[buf], lane, j), o[j]);
    if (more) {
      wait_async();
      __syncthreads();
      buf ^= 1;
    }
  }
#pragma unroll
  for (int r = 0; r < 8; ++r) {
    float inv = 1.f / rowl[r];
    size_t row = (size_t)b * N_ + qt * 16 + r + 8 * hi;
#pragma unroll
    for (int j = 0; j < 4; ++j)
      out16[row * DIM + j * 16 + nn] = (_Float16)(o[j][r] * inv);
  }
}

// ---------------- local branch prep ----------------
__global__ void k_dw3x3(const float* __restrict__ x, const float* __restrict__ w,
                        const float* __restrict__ bias, _Float16* __restrict__ outp) {
  size_t i = (size_t)blockIdx.x * 256 + threadIdx.x;
  if (i >= (size_t)BN * DL) return;
  int c = (int)(i % DL);
  size_t bn = i / DL;
  int n = (int)(bn % N_); int b = (int)(bn / N_);
  int hh = n / W_, ww = n % W_;
  float acc = bias[c];
#pragma unroll
  for (int ky = 0; ky < 3; ++ky) {
    int y = hh + ky - 1; if (y < 0 || y >= H_) continue;
#pragma unroll
    for (int kx = 0; kx < 3; ++kx) {
      int xx = ww + kx - 1; if (xx < 0 || xx >= W_) continue;
      acc += x[((size_t)b * N_ + y * W_ + xx) * DIM + D0 + c] * w[c * 9 + ky * 3 + kx];
    }
  }
  outp[i] = (_Float16)acc;
}

// exact reference semantics of the (L*hd)->(hd,L) reshape + mean over L, with reflect pad
__global__ void k_window_mean(const float* __restrict__ kvl, float* __restrict__ outm,
                              int cbase, int stride, int dil, int ngrid) {
  int i = blockIdx.x * 256 + threadIdx.x;
  int total = BATCH * NHD * 49 * HD;
  if (i >= total) return;
  int d2 = i % HD; int t = i / HD;
  int pos = t % 49; t /= 49;
  int h = t % NHD; int b = t / NHD;
  int pi = pos / 7, pj = pos % 7;
  int L = ngrid * ngrid;
  float acc = 0.f;
  for (int l2 = 0; l2 < L; ++l2) {
    int flat = d2 * L + l2;
    int l = flat / HD, d = flat % HD;
    int wi = l / ngrid, wj = l % ngrid;
    int y = wi * stride + pi * dil; if (y >= H_) y = 2 * (H_ - 1) - y;
    int xx = wj * stride + pj * dil; if (xx >= W_) xx = 2 * (W_ - 1) - xx;
    acc += kvl[((size_t)b * N_ + y * W_ + xx) * DL + cbase + h * HD + d];
  }
  outm[i] = acc / (float)L;
}

__global__ void k_fc_small(const float* __restrict__ in, const float* __restrict__ w,
                           float* __restrict__ outp, int rows, int K, int Ncol) {
  int i = blockIdx.x * 256 + threadIdx.x;
  if (i >= rows * Ncol) return;
  int c = i % Ncol, r = i / Ncol;
  float acc = 0.f;
  for (int k = 0; k < K; ++k) acc += in[r * K + k] * w[k * Ncol + c];
  outp[i] = acc;
}

// K^T swizzled: block ((b*8+h)*4 + keytile j), elem [(key%16)*32 + d]
__global__ void k_pack_kt(const float* __restrict__ kvh, _Float16* __restrict__ kt) {
  int i = blockIdx.x * 256 + threadIdx.x;
  if (i >= BATCH * NHD * 4 * 512) return;
  int half = i & 511; int blk = i >> 9;
  int j = blk & 3; int bh = blk >> 2;
  int n = half >> 5, d = half & 31;
  int key = j * 16 + n;
  _Float16 v = (_Float16)0.f;
  if (d < HD && key < 49) v = (_Float16)kvh[((size_t)bh * 49 + key) * 56 + d];
  kt[i] = v;
}
// V swizzled: block ((b*8+h)*4 + keychunk*2 + dimtile), elem [(d%16)*32 + key%32]
__global__ void k_pack_v(const float* __restrict__ kvh, const float* __restrict__ lcw,
                         const float* __restrict__ lcb, _Float16* __restrict__ vp) {
  int i = blockIdx.x * 256 + threadIdx.x;
  if (i >= BATCH * NHD * 4 * 512) return;
  int half = i & 511; int blk = i >> 9;
  int j = blk & 1; int kt2 = (blk >> 1) & 1; int bh = blk >> 2;
  int n = half >> 5, kk = half & 31;
  int key = kt2 * 32 + kk, d = j * 16 + n;
  _Float16 res = (_Float16)0.f;
  if (d < HD && key < 49) {
    int h = bh % NHD;
    int c = h * HD + d;
    int pi = key / 7, pj = key % 7;
    float acc = lcb[c];
#pragma unroll
    for (int ky = 0; ky < 3; ++ky) {
      int y = pi + ky - 1; if (y < 0 || y >= 7) continue;
#pragma unroll
      for (int kx = 0; kx < 3; ++kx) {
        int xx = pj + kx - 1; if (xx < 0 || xx >= 7) continue;
        acc += kvh[((size_t)bh * 49 + y * 7 + xx) * 56 + 28 + d] * lcw[c * 9 + ky * 3 + kx];
      }
    }
    res = (_Float16)(kvh[((size_t)bh * 49 + key) * 56 + 28 + d] + acc);
  }
  vp[i] = res;
}

// ---------------- local attention: 1 wave / (b, head, 16 queries) ----------------
__global__ void k_local_attn(const float* __restrict__ q, const _Float16* __restrict__ ktg_,
                             const _Float16* __restrict__ vvg_, _Float16* __restrict__ out16,
                             int branch) {
  __shared__ __align__(16) _Float16 qs[16 * 32];
  __shared__ __align__(16) _Float16 kts[4 * 512];
  __shared__ __align__(16) _Float16 vs[4 * 512];
  __shared__ __align__(16) _Float16 ps[16 * 64];
  int lane = threadIdx.x;
  int tile = blockIdx.x;
  int ntile = N_ / 16;
  int b = tile / (NHD * ntile);
  int rem = tile % (NHD * ntile);
  int h = rem / ntile, qt = rem % ntile;
  const float scale = 0.18898223650461363f; // 28^-0.5
  const _Float16* ksrc = ktg_ + ((size_t)(b * NHD + h) * 4) * 512;
  const _Float16* vsrc = vvg_ + ((size_t)(b * NHD + h) * 4) * 512;
#pragma unroll
  for (int it = 0; it < 8; ++it) {
    int i = lane + it * 32;
    async_copy_b128((unsigned)(size_t)&kts[i * 8], ksrc + i * 8);
    async_copy_b128((unsigned)(size_t)&vs[i * 8],  vsrc + i * 8);
  }
#pragma unroll
  for (int it = 0; it < 16; ++it) {
    int i = lane + it * 32;
    int r = i >> 5, d = i & 31;
    _Float16 v = (_Float16)0.f;
    if (d < HD)
      v = (_Float16)(q[((size_t)b * N_ + qt * 16 + r) * DIM + D0 + h * 56 + branch * HD + d] * scale);
    qs[i] = v;
  }
  wait_async();
  __syncthreads();
  v16h a = load_a_frag(qs, 32, lane, 0);
  const v8f vz = {0.f,0.f,0.f,0.f,0.f,0.f,0.f,0.f};
  v8f s[4];
#pragma unroll
  for (int j = 0; j < 4; ++j)
    s[j] = WMMA_F16(a, load_bswz_frag(kts, lane, j), vz);
  int nn = lane & 15, hi = lane >> 4;
#pragma unroll
  for (int j = 0; j < 4; ++j) {
    int key = j * 16 + nn;
    if (key >= 49) {
#pragma unroll
      for (int r = 0; r < 8; ++r) s[j][r] = -1e30f;
    }
  }
  float rowl[8];
#pragma unroll
  for (int r = 0; r < 8; ++r) {
    float mx = fmaxf(fmaxf(s[0][r], s[1][r]), fmaxf(s[2][r], s[3][r]));
    for (int d = 1; d < 16; d <<= 1) mx = fmaxf(mx, __shfl_xor(mx, d, 16));
    float rs = 0.f;
#pragma unroll
    for (int j = 0; j < 4; ++j) { float p = __expf(s[j][r] - mx); s[j][r] = p; rs += p; }
    for (int d = 1; d < 16; d <<= 1) rs += __shfl_xor(rs, d, 16);
    rowl[r] = rs;
  }
#pragma unroll
  for (int r = 0; r < 8; ++r)
#pragma unroll
    for (int j = 0; j < 4; ++j)
      ps[(r + 8 * hi) * 64 + j * 16 + nn] = (_Float16)s[j][r];
  __syncthreads();
  v16h pa0 = load_a_frag(ps, 64, lane, 0);
  v16h pa1 = load_a_frag(ps, 64, lane, 32);
  v8f o0 = vz, o1 = vz;
  o0 = WMMA_F16(pa0, load_bswz_frag(vs, lane, 0), o0);  // keys 0-31, dims 0-15
  o0 = WMMA_F16(pa1, load_bswz_frag(vs, lane, 2), o0);  // keys 32-63, dims 0-15
  o1 = WMMA_F16(pa0, load_bswz_frag(vs, lane, 1), o1);  // keys 0-31, dims 16-31
  o1 = WMMA_F16(pa1, load_bswz_frag(vs, lane, 3), o1);
  int cb = D0 + branch * 224 + h * HD;
#pragma unroll
  for (int r = 0; r < 8; ++r) {
    float inv = 1.f / rowl[r];
    size_t row = (size_t)b * N_ + qt * 16 + r + 8 * hi;
    out16[row * DIM + cb + nn] = (_Float16)(o0[r] * inv);
    if (16 + nn < HD) out16[row * DIM + cb + 16 + nn] = (_Float16)(o1[r] * inv);
  }
}

// ---------------- launch ----------------
extern "C" void kernel_launch(void* const* d_in, const int* in_sizes, int n_in,
                              void* d_out, int out_size, void* d_ws, size_t ws_size,
                              hipStream_t stream) {
  (void)in_sizes; (void)n_in; (void)out_size; (void)ws_size;
  const float* x      = (const float*)d_in[0];
  const float* Wq     = (const float*)d_in[1];
  const float* Wkv_g  = (const float*)d_in[2];
  const float* sr_w   = (const float*)d_in[3];
  const float* sr_b   = (const float*)d_in[4];
  const float* ln_g   = (const float*)d_in[5];
  const float* ln_b   = (const float*)d_in[6];
  const float* lcg_w  = (const float*)d_in[7];
  const float* lcg_b  = (const float*)d_in[8];
  const float* kvl_dw = (const float*)d_in[9];
  const float* kvl_db = (const float*)d_in[10];
  const float* kvl_pw = (const float*)d_in[11];
  const float* kvl_pb = (const float*)d_in[12];
  const float* fcw[2] = {(const float*)d_in[13], (const float*)d_in[17]};
  const float* shw[2] = {(const float*)d_in[14], (const float*)d_in[18]};
  const float* lcw[2] = {(const float*)d_in[15], (const float*)d_in[19]};
  const float* lcb[2] = {(const float*)d_in[16], (const float*)d_in[20]};
  const float* proj_w = (const float*)d_in[21];
  const float* proj_b = (const float*)d_in[22];
  float* out = (float*)d_out;

  char* ws = (char*)d_ws;
  size_t off = 0;
  auto alloc = [&](size_t bytes) -> char* {
    char* p = ws + off;
    off = (off + bytes + 255) & ~(size_t)255;
    return p;
  };
  _Float16* x16    = (_Float16*)alloc((size_t)BN * DIM * 2);
  _Float16* wq16   = (_Float16*)alloc((size_t)DIM * DIM * 2);
  float*    qbuf   = (float*)   alloc((size_t)BN * DIM * 4);
  _Float16* xg16   = (_Float16*)alloc((size_t)BATCH * NR * D0 * 2);
  _Float16* wkvg16 = (_Float16*)alloc((size_t)D0 * 128 * 2);
  float*    kvg    = (float*)   alloc((size_t)BATCH * NR * 128 * 4);
  _Float16* kg16   = (_Float16*)alloc((size_t)BATCH * NR * D0 * 2);
  _Float16* vg16   = (_Float16*)alloc((size_t)BATCH * NR * D0 * 2);
  _Float16* dw16   = (_Float16*)alloc((size_t)BN * DL * 2);
  _Float16* pwT16  = (_Float16*)alloc((size_t)DL * DL * 2);
  float*    kvl    = (float*)   alloc((size_t)BN * DL * 4);
  float*    kvs[2], *t1[2], *kvh[2];
  _Float16 *kt16[2], *v16b[2];
  for (int br = 0; br < 2; ++br) {
    kvs[br]  = (float*)alloc((size_t)BATCH * NHD * 49 * HD * 4);
    t1[br]   = (float*)alloc((size_t)BATCH * NHD * 49 * HD * 4);
    kvh[br]  = (float*)alloc((size_t)BATCH * NHD * 49 * 56 * 4);
    kt16[br] = (_Float16*)alloc((size_t)BATCH * NHD * 4 * 512 * 2);
    v16b[br] = (_Float16*)alloc((size_t)BATCH * NHD * 4 * 512 * 2);
  }
  _Float16* out16  = (_Float16*)alloc((size_t)BN * DIM * 2);
  _Float16* pj16   = (_Float16*)alloc((size_t)DIM * DIM * 2);

  auto g1 = [](long n) { return dim3((unsigned)((n + 255) / 256)); };

  // q = x @ Wq (f16 WMMA)
  k_cvt<<<g1((long)BN * DIM), 256, 0, stream>>>(x, x16, (long)BN * DIM);
  k_cvt_swz<<<g1((long)DIM * DIM), 256, 0, stream>>>(Wq, wq16, DIM, DIM, 0);
  k_gemm_wmma<<<dim3(BN / 32, DIM / 64), 32, 0, stream>>>(x16, wq16, nullptr, qbuf, BN, DIM, DIM);

  // global branch
  k_sr_ln<<<BATCH * NR, 64, 0, stream>>>(x, sr_w, sr_b, ln_g, ln_b, xg16);
  k_cvt_swz<<<g1((long)D0 * 128), 256, 0, stream>>>(Wkv_g, wkvg16, D0, 128, 0);
  k_gemm_wmma<<<dim3(BATCH * NR / 32, 128 / 64), 32, 0, stream>>>(xg16, wkvg16, nullptr, kvg,
                                                                   BATCH * NR, 128, D0);
  k_finalize_kvg<<<g1((long)BATCH * NR * D0), 256, 0, stream>>>(kvg, lcg_w, lcg_b, kg16, vg16);
  k_flash_global<<<BATCH * (N_ / 16), 32, 0, stream>>>(qbuf, kg16, vg16, out16);

  // local kv: depthwise + pointwise (as WMMA GEMM, weights pre-transposed+swizzled)
  k_dw3x3<<<g1((long)BN * DL), 256, 0, stream>>>(x, kvl_dw, kvl_db, dw16);
  k_cvt_swz<<<g1((long)DL * DL), 256, 0, stream>>>(kvl_pw, pwT16, DL, DL, 1);
  k_gemm_wmma<<<dim3(BN / 32, DL / 64), 32, 0, stream>>>(dw16, pwT16, kvl_pb, kvl, BN, DL, DL);

  // local branches
  const int strides[2] = {7, 13};
  const int dils[2]    = {1, 2};
  const int ngrids[2]  = {16, 9};
  for (int br = 0; br < 2; ++br) {
    k_window_mean<<<g1((long)BATCH * NHD * 49 * HD), 256, 0, stream>>>(
        kvl, kvs[br], br * 224, strides[br], dils[br], ngrids[br]);
    int rows = BATCH * NHD * 49;
    k_fc_small<<<g1((long)rows * HD), 256, 0, stream>>>(kvs[br], fcw[br], t1[br], rows, HD, HD);
    k_fc_small<<<g1((long)rows * 56), 256, 0, stream>>>(t1[br], shw[br], kvh[br], rows, HD, 56);
    k_pack_kt<<<g1((long)BATCH * NHD * 4 * 512), 256, 0, stream>>>(kvh[br], kt16[br]);
    k_pack_v<<<g1((long)BATCH * NHD * 4 * 512), 256, 0, stream>>>(kvh[br], lcw[br], lcb[br], v16b[br]);
    k_local_attn<<<BATCH * NHD * (N_ / 16), 32, 0, stream>>>(qbuf, kt16[br], v16b[br], out16, br);
  }

  // final projection (f16 WMMA, fp32 out + bias)
  k_cvt_swz<<<g1((long)DIM * DIM), 256, 0, stream>>>(proj_w, pj16, DIM, DIM, 0);
  k_gemm_wmma<<<dim3(BN / 32, DIM / 64), 32, 0, stream>>>(out16, pj16, proj_b, out, BN, DIM, DIM);
}
